// VASNet_Audio128_Att_59296318489037
// MI455X (gfx1250) — matrix operand order repaired and checked by reading, not compiled, not run on hardware
//
#include <hip/hip_runtime.h>
#include <hip/hip_bf16.h>

// ---------------------------------------------------------------------------
// Problem constants (fixed by the reference)
// ---------------------------------------------------------------------------
#define SEQ      6144
#define DVID     1024
#define DAUD     128
#define DCOMB    1152
#define DHID     1024
#define APERTURE 20
#define LN_EPS   1e-6f

typedef __attribute__((ext_vector_type(16))) __bf16 v16bf;
typedef __attribute__((ext_vector_type(8)))  float  v8f;

// ---------------------------------------------------------------------------
// f32 -> bf16 (round to nearest even), grid-stride
// ---------------------------------------------------------------------------
__global__ void f32_to_bf16_kernel(const float* __restrict__ src,
                                   unsigned short* __restrict__ dst, size_t n) {
    size_t stride = (size_t)gridDim.x * blockDim.x;
    for (size_t i = (size_t)blockIdx.x * blockDim.x + threadIdx.x; i < n; i += stride) {
        unsigned u = __float_as_uint(src[i]);
        unsigned r = (u + 0x7fffu + ((u >> 16) & 1u)) >> 16;
        dst[i] = (unsigned short)r;
    }
}

__device__ __forceinline__ unsigned short f2bf(float f) {
    unsigned u = __float_as_uint(f);
    return (unsigned short)((u + 0x7fffu + ((u >> 16) & 1u)) >> 16);
}

// ---------------------------------------------------------------------------
// Zero-fill (float4 stores), grid-stride
// ---------------------------------------------------------------------------
__global__ void zero_f32x4_kernel(float4* __restrict__ p, size_t n4) {
    size_t stride = (size_t)gridDim.x * blockDim.x;
    float4 z; z.x = 0.f; z.y = 0.f; z.z = 0.f; z.w = 0.f;
    for (size_t i = (size_t)blockIdx.x * blockDim.x + threadIdx.x; i < n4; i += stride)
        p[i] = z;
}

// ---------------------------------------------------------------------------
// WMMA bf16 GEMM:  C[M,Nout] = relu?( alpha * (A @ W^T) + bias )
//   A: bf16 [M,K] row-major,  W: bf16 [Nout,K] row-major (torch Linear W)
//   Block = 256 threads (8 waves, wave32); block tile 128x128; wave tile 32x64
//   (2x4 grid of v_wmma_f32_16x16x32_bf16 accumulators); K-step 32.
//   Double-buffered LDS: one s_barrier per K-step; global_prefetch 2 steps out.
//   Requires: M%128==0, Nout%128==0, K%32==0.
// ---------------------------------------------------------------------------
#define GEMM_BM 128
#define GEMM_BN 128
#define GEMM_BK 32
#define LDS_STRIDE 40   // bf16 elems per LDS row: 32 data + 8 pad (80B, 16B-aligned)

__global__ __launch_bounds__(256)
void gemm_bf16_wmma_kernel(const unsigned short* __restrict__ A,
                           const unsigned short* __restrict__ W,
                           float* __restrict__ C,
                           int M, int Nout, int K,
                           float alpha, const float* __restrict__ bias, int relu) {
    __shared__ __align__(16) unsigned short As[2][GEMM_BM * LDS_STRIDE];
    __shared__ __align__(16) unsigned short Bs[2][GEMM_BN * LDS_STRIDE];

    const int t    = threadIdx.x;
    const int lane = t & 31;
    const int wave = t >> 5;
    const int wm   = wave >> 1;      // 0..3 : 32-row band inside block tile
    const int wn   = wave & 1;       // 0..1 : 64-col band inside block tile
    const int m0   = blockIdx.x * GEMM_BM;
    const int n0   = blockIdx.y * GEMM_BN;
    const int half = lane >> 4;      // 0/1
    const int lrow = lane & 15;

    // cooperative load geometry: 128 rows x 4 (16B) chunks = 512 chunks,
    // thread t covers (row r0, chunk q0) and (row r0+64, chunk q0).
    const int r0 = t >> 2, q0 = t & 3;
    const uint4* Ag0 = reinterpret_cast<const uint4*>(A + (size_t)(m0 + r0) * K);
    const uint4* Ag1 = reinterpret_cast<const uint4*>(A + (size_t)(m0 + r0 + 64) * K);
    const uint4* Bg0 = reinterpret_cast<const uint4*>(W + (size_t)(n0 + r0) * K);
    const uint4* Bg1 = reinterpret_cast<const uint4*>(W + (size_t)(n0 + r0 + 64) * K);

    uint4 ra0, ra1, rb0, rb1;
    auto gload = [&](int k0) {
        int ko = (k0 >> 3) + q0;          // uint4 index along the row
        ra0 = Ag0[ko]; ra1 = Ag1[ko];
        rb0 = Bg0[ko]; rb1 = Bg1[ko];
    };
    auto lstore = [&](int buf) {
        *reinterpret_cast<uint4*>(&As[buf][(r0)      * LDS_STRIDE + q0 * 8]) = ra0;
        *reinterpret_cast<uint4*>(&As[buf][(r0 + 64) * LDS_STRIDE + q0 * 8]) = ra1;
        *reinterpret_cast<uint4*>(&Bs[buf][(r0)      * LDS_STRIDE + q0 * 8]) = rb0;
        *reinterpret_cast<uint4*>(&Bs[buf][(r0 + 64) * LDS_STRIDE + q0 * 8]) = rb1;
    };

    v8f acc[2][4] = {};

    gload(0);
    lstore(0);
    __syncthreads();

    const int nt = K / GEMM_BK;
    for (int it = 0; it < nt; ++it) {
        const int buf = it & 1;

        if (it + 1 < nt) gload((it + 1) * GEMM_BK);          // stage next tile
        if (it + 2 < nt) {                                   // prefetch tile+2
            int kp = (it + 2) * GEMM_BK;
            __builtin_prefetch(A + (size_t)(m0 + (t & 127)) * K + kp, 0, 3);
            __builtin_prefetch(W + (size_t)(n0 + (t & 127)) * K + kp, 0, 3);
        }

        // A fragments: 16x32 bf16, lane = row M=lrow; chunks at bytes h*16, 32+h*16
        v16bf afrag[2];
#pragma unroll
        for (int mi = 0; mi < 2; ++mi) {
            const uint4* p = reinterpret_cast<const uint4*>(
                &As[buf][(wm * 32 + mi * 16 + lrow) * LDS_STRIDE]);
            reinterpret_cast<uint4*>(&afrag[mi])[0] = p[half];
            reinterpret_cast<uint4*>(&afrag[mi])[1] = p[2 + half];
        }
        // B fragments: 32x16 bf16, lane = col N=lrow; chunks at bytes h*16, 32+h*16
        v16bf bfrag[4];
#pragma unroll
        for (int ni = 0; ni < 4; ++ni) {
            const uint4* p = reinterpret_cast<const uint4*>(
                &Bs[buf][(wn * 64 + ni * 16 + lrow) * LDS_STRIDE]);
            reinterpret_cast<uint4*>(&bfrag[ni])[0] = p[half];
            reinterpret_cast<uint4*>(&bfrag[ni])[1] = p[2 + half];
        }

#pragma unroll
        for (int mi = 0; mi < 2; ++mi)
#pragma unroll
            for (int ni = 0; ni < 4; ++ni)
                acc[mi][ni] = __builtin_amdgcn_wmma_f32_16x16x32_bf16(
                    false, afrag[mi], false, bfrag[ni], (short)0, acc[mi][ni],
                    false, false);

        if (it + 1 < nt) {
            lstore(buf ^ 1);          // safe: buf^1 last read before prior barrier
            __syncthreads();
        }
    }

    // epilogue: C/D layout is VGPR d <-> row (half*8 + d), col = lrow
#pragma unroll
    for (int mi = 0; mi < 2; ++mi)
#pragma unroll
        for (int ni = 0; ni < 4; ++ni) {
            int col   = n0 + wn * 64 + ni * 16 + lrow;
            float bv  = bias ? bias[col] : 0.f;
            int rbase = m0 + wm * 32 + mi * 16 + half * 8;
#pragma unroll
            for (int dd = 0; dd < 8; ++dd) {
                float v = acc[mi][ni][dd] * alpha + bv;
                if (relu) v = fmaxf(v, 0.f);
                C[(size_t)(rbase + dd) * Nout + col] = v;
            }
        }
}

// ---------------------------------------------------------------------------
// Banded softmax: one block per row i. logits[j] = dot(Q[i], K[j]) for
// |i-j| < APERTURE (<=39 entries). Writes normalized att row straight into
// d_out's att_comb region (rest pre-zeroed).
// ---------------------------------------------------------------------------
__global__ __launch_bounds__(256)
void band_softmax_kernel(const float* __restrict__ Q, const float* __restrict__ Kc,
                         float* __restrict__ att, long ldatt, int coloff,
                         int d, int n) {
    __shared__ float sl[64];
    const int i = blockIdx.x;
    const int t = threadIdx.x, wave = t >> 5, lane = t & 31;
    int jlo = i - (APERTURE - 1); if (jlo < 0) jlo = 0;
    int jhi = i + (APERTURE - 1); if (jhi > n - 1) jhi = n - 1;
    const int nj = jhi - jlo + 1;

    for (int jj = wave; jj < nj; jj += 8) {
        int j = jlo + jj;
        const float* qp = Q  + (size_t)i * d;
        const float* kp = Kc + (size_t)j * d;
        float s = 0.f;
        for (int c = lane; c < d; c += 32) s += qp[c] * kp[c];
        for (int o = 16; o; o >>= 1) s += __shfl_xor(s, o, 32);
        if (lane == 0) sl[jj] = s;
    }
    __syncthreads();

    if (wave == 0) {
        float m = -3.4e38f;
        for (int jj = lane; jj < nj; jj += 32) m = fmaxf(m, sl[jj]);
        for (int o = 16; o; o >>= 1) m = fmaxf(m, __shfl_xor(m, o, 32));
        float e[2]; float s = 0.f;
#pragma unroll
        for (int u = 0; u < 2; ++u) {
            int jj = lane + u * 32;
            e[u] = (jj < nj) ? expf(sl[jj] - m) : 0.f;
            s += e[u];
        }
        for (int o = 16; o; o >>= 1) s += __shfl_xor(s, o, 32);
        float inv = 1.f / s;
#pragma unroll
        for (int u = 0; u < 2; ++u) {
            int jj = lane + u * 32;
            if (jj < nj) sl[jj] = e[u] * inv;
        }
    }
    __syncthreads();

    for (int jj = t; jj < nj; jj += 256)
        att[(size_t)i * ldatt + coloff + jlo + jj] = sl[jj];
}

// ---------------------------------------------------------------------------
// y = att^T @ V  (banded):  Y[j,:] = sum_{|i-j|<AP} att[i][j] * V[i,:]
// One block per output row j; <=4 channels per thread.
// ---------------------------------------------------------------------------
__global__ __launch_bounds__(256)
void att_t_v_kernel(const float* __restrict__ att, long ldatt, int coloff,
                    const float* __restrict__ V, float* __restrict__ Y,
                    int d, int n) {
    const int j = blockIdx.x, t = threadIdx.x;
    int ilo = j - (APERTURE - 1); if (ilo < 0) ilo = 0;
    int ihi = j + (APERTURE - 1); if (ihi > n - 1) ihi = n - 1;
    float acc[4] = {0.f, 0.f, 0.f, 0.f};
    for (int i = ilo; i <= ihi; ++i) {
        float w = att[(size_t)i * ldatt + coloff + j];
        const float* vp = V + (size_t)i * d;
#pragma unroll
        for (int u = 0; u < 4; ++u) {
            int c = t + u * 256;
            if (c < d) acc[u] += w * vp[c];
        }
    }
#pragma unroll
    for (int u = 0; u < 4; ++u) {
        int c = t + u * 256;
        if (c < d) Y[(size_t)j * d + c] = acc[u];
    }
}

// ---------------------------------------------------------------------------
// LayerNorm over dim d (one block per row): out = LN(in + res) * g + b
// Optional f32 output (stride ldf) and/or bf16 output (stride ldb, col off)
// ---------------------------------------------------------------------------
__global__ __launch_bounds__(256)
void ln_kernel(const float* __restrict__ in, const float* __restrict__ res,
               const float* __restrict__ g, const float* __restrict__ bta,
               float* __restrict__ outf, int ldf,
               unsigned short* __restrict__ outb, int ldb, int offb, int d) {
    __shared__ float red[8];
    __shared__ float bc;
    const int row = blockIdx.x, t = threadIdx.x, wave = t >> 5, lane = t & 31;

    float v[4];
    float s = 0.f;
#pragma unroll
    for (int u = 0; u < 4; ++u) {
        int c = t + u * 256;
        float x = 0.f;
        if (c < d) {
            x = in[(size_t)row * d + c];
            if (res) x += res[(size_t)row * d + c];
        }
        v[u] = x;
        s += x;
    }
    for (int o = 16; o; o >>= 1) s += __shfl_xor(s, o, 32);
    if (lane == 0) red[wave] = s;
    __syncthreads();
    if (t == 0) { float tot = 0.f; for (int w = 0; w < 8; ++w) tot += red[w]; bc = tot; }
    __syncthreads();
    const float mu = bc / d;

    float s2 = 0.f;
#pragma unroll
    for (int u = 0; u < 4; ++u) {
        int c = t + u * 256;
        if (c < d) { float dv = v[u] - mu; s2 += dv * dv; }
    }
    for (int o = 16; o; o >>= 1) s2 += __shfl_xor(s2, o, 32);
    __syncthreads();                       // red reuse safe: t0 done reading
    if (lane == 0) red[wave] = s2;
    __syncthreads();
    if (t == 0) { float tot = 0.f; for (int w = 0; w < 8; ++w) tot += red[w]; bc = tot; }
    __syncthreads();
    const float rstd = rsqrtf(bc / d + LN_EPS);

#pragma unroll
    for (int u = 0; u < 4; ++u) {
        int c = t + u * 256;
        if (c < d) {
            float o = (v[u] - mu) * rstd * g[c] + bta[c];
            if (outf) outf[(size_t)row * ldf + c] = o;
            if (outb) outb[(size_t)row * ldb + offb + c] = f2bf(o);
        }
    }
}

// ---------------------------------------------------------------------------
// score[i] = sigmoid(dot(h[i], w) + b0)   (one block per row, d = 1024)
// ---------------------------------------------------------------------------
__global__ __launch_bounds__(256)
void score_kernel(const float* __restrict__ h, const float* __restrict__ w,
                  const float* __restrict__ b, float* __restrict__ out, int d) {
    __shared__ float red[8];
    const int row = blockIdx.x, t = threadIdx.x, wave = t >> 5, lane = t & 31;
    float s = 0.f;
    for (int c = t; c < d; c += 256) s += h[(size_t)row * d + c] * w[c];
    for (int o = 16; o; o >>= 1) s += __shfl_xor(s, o, 32);
    if (lane == 0) red[wave] = s;
    __syncthreads();
    if (t == 0) {
        float tot = 0.f;
        for (int wv = 0; wv < 8; ++wv) tot += red[wv];
        tot += b[0];
        out[row] = 1.f / (1.f + expf(-tot));
    }
}

// ---------------------------------------------------------------------------
// Orchestration
// ---------------------------------------------------------------------------
extern "C" void kernel_launch(void* const* d_in, const int* in_sizes, int n_in,
                              void* d_out, int out_size, void* d_ws, size_t ws_size,
                              hipStream_t stream) {
    (void)in_sizes; (void)n_in; (void)out_size; (void)ws_size;

    const float* x    = (const float*)d_in[0];   // [SEQ, 1024]
    const float* xa   = (const float*)d_in[1];   // [SEQ, 128]
    const float* Wk_v = (const float*)d_in[4];
    const float* Wq_v = (const float*)d_in[5];
    const float* Wv_v = (const float*)d_in[6];
    const float* Wo_v = (const float*)d_in[7];
    const float* Wk_a = (const float*)d_in[8];
    const float* Wq_a = (const float*)d_in[9];
    const float* Wv_a = (const float*)d_in[10];
    const float* Wo_a = (const float*)d_in[11];
    const float* ka_w = (const float*)d_in[12];  // [1024, 1152]
    const float* ka_b = (const float*)d_in[13];
    const float* kd_w = (const float*)d_in[14];  // [1024]
    const float* kd_b = (const float*)d_in[15];
    const float* ln_y_g  = (const float*)d_in[16];
    const float* ln_y_b  = (const float*)d_in[17];
    const float* ln_ya_g = (const float*)d_in[18];
    const float* ln_ya_b = (const float*)d_in[19];
    const float* ln_ka_g = (const float*)d_in[20];
    const float* ln_ka_b = (const float*)d_in[21];

    float* out = (float*)d_out;          // [SEQ] score
    float* att = out + SEQ;              // [SEQ, 2*SEQ] att_comb
    const long ldatt = 2L * SEQ;

    // --- workspace carve-out (aligned 256B) ---
    char* ws = (char*)d_ws;
    size_t off = 0;
    auto alloc = [&](size_t bytes) -> void* {
        void* p = ws + off;
        off = (off + bytes + 255) & ~(size_t)255;
        return p;
    };
    unsigned short* x_bf   = (unsigned short*)alloc((size_t)SEQ * DVID * 2);
    unsigned short* xa_bf  = (unsigned short*)alloc((size_t)SEQ * DAUD * 2);
    unsigned short* wkv_bf = (unsigned short*)alloc((size_t)DVID * DVID * 2);
    unsigned short* wqv_bf = (unsigned short*)alloc((size_t)DVID * DVID * 2);
    unsigned short* wvv_bf = (unsigned short*)alloc((size_t)DVID * DVID * 2);
    unsigned short* wov_bf = (unsigned short*)alloc((size_t)DVID * DVID * 2);
    unsigned short* wka_bf = (unsigned short*)alloc((size_t)DAUD * DAUD * 2);
    unsigned short* wqa_bf = (unsigned short*)alloc((size_t)DAUD * DAUD * 2);
    unsigned short* wva_bf = (unsigned short*)alloc((size_t)DAUD * DAUD * 2);
    unsigned short* woa_bf = (unsigned short*)alloc((size_t)DAUD * DAUD * 2);
    unsigned short* kaw_bf = (unsigned short*)alloc((size_t)DHID * DCOMB * 2);
    float* Qv = (float*)alloc((size_t)SEQ * DVID * 4);
    float* Kv = (float*)alloc((size_t)SEQ * DVID * 4);
    float* Vv = (float*)alloc((size_t)SEQ * DVID * 4);
    float* Qa = (float*)alloc((size_t)SEQ * DAUD * 4);
    float* Ka = (float*)alloc((size_t)SEQ * DAUD * 4);
    float* Va = (float*)alloc((size_t)SEQ * DAUD * 4);
    float* Yv = (float*)alloc((size_t)SEQ * DVID * 4);
    float* Ya = (float*)alloc((size_t)SEQ * DAUD * 4);
    unsigned short* ycomb_bf = (unsigned short*)alloc((size_t)SEQ * DCOMB * 2);
    // deterministic buffer reuse after their producers are consumed:
    float* Yv2 = Qv;                     // Wo_v output
    float* Ya2 = Qa;                     // Wo_a output
    unsigned short* yv_bf = x_bf;        // bf16 of Yv
    unsigned short* ya_bf = xa_bf;       // bf16 of Ya
    float* H   = Kv;                     // MLP hidden (relu)
    float* HLn = Vv;                     // LN(hidden)

    const dim3 blk(256);
    const dim3 gV(SEQ / GEMM_BM, DVID / GEMM_BN);   // (48, 8)
    const dim3 gA(SEQ / GEMM_BM, DAUD / GEMM_BN);   // (48, 1)

    // 1) f32 -> bf16 conversions
    f32_to_bf16_kernel<<<1024, blk, 0, stream>>>(x,    x_bf,   (size_t)SEQ * DVID);
    f32_to_bf16_kernel<<<256,  blk, 0, stream>>>(xa,   xa_bf,  (size_t)SEQ * DAUD);
    f32_to_bf16_kernel<<<512,  blk, 0, stream>>>(Wk_v, wkv_bf, (size_t)DVID * DVID);
    f32_to_bf16_kernel<<<512,  blk, 0, stream>>>(Wq_v, wqv_bf, (size_t)DVID * DVID);
    f32_to_bf16_kernel<<<512,  blk, 0, stream>>>(Wv_v, wvv_bf, (size_t)DVID * DVID);
    f32_to_bf16_kernel<<<512,  blk, 0, stream>>>(Wo_v, wov_bf, (size_t)DVID * DVID);
    f32_to_bf16_kernel<<<32,   blk, 0, stream>>>(Wk_a, wka_bf, (size_t)DAUD * DAUD);
    f32_to_bf16_kernel<<<32,   blk, 0, stream>>>(Wq_a, wqa_bf, (size_t)DAUD * DAUD);
    f32_to_bf16_kernel<<<32,   blk, 0, stream>>>(Wv_a, wva_bf, (size_t)DAUD * DAUD);
    f32_to_bf16_kernel<<<32,   blk, 0, stream>>>(Wo_a, woa_bf, (size_t)DAUD * DAUD);
    f32_to_bf16_kernel<<<512,  blk, 0, stream>>>(ka_w, kaw_bf, (size_t)DHID * DCOMB);

    // 2) QKV projections (alpha=0.06 folded into Q)
    gemm_bf16_wmma_kernel<<<gV, blk, 0, stream>>>(x_bf, wkv_bf, Kv, SEQ, DVID, DVID, 1.0f,  nullptr, 0);
    gemm_bf16_wmma_kernel<<<gV, blk, 0, stream>>>(x_bf, wqv_bf, Qv, SEQ, DVID, DVID, 0.06f, nullptr, 0);
    gemm_bf16_wmma_kernel<<<gV, blk, 0, stream>>>(x_bf, wvv_bf, Vv, SEQ, DVID, DVID, 1.0f,  nullptr, 0);
    gemm_bf16_wmma_kernel<<<gA, blk, 0, stream>>>(xa_bf, wka_bf, Ka, SEQ, DAUD, DAUD, 1.0f,  nullptr, 0);
    gemm_bf16_wmma_kernel<<<gA, blk, 0, stream>>>(xa_bf, wqa_bf, Qa, SEQ, DAUD, DAUD, 0.06f, nullptr, 0);
    gemm_bf16_wmma_kernel<<<gA, blk, 0, stream>>>(xa_bf, wva_bf, Va, SEQ, DAUD, DAUD, 1.0f,  nullptr, 0);

    // 3) zero att_comb then fill band (softmax rows go straight into d_out)
    zero_f32x4_kernel<<<8192, blk, 0, stream>>>((float4*)att, (size_t)SEQ * ldatt / 4);
    band_softmax_kernel<<<SEQ, blk, 0, stream>>>(Qv, Kv, att, ldatt, 0,   DVID, SEQ);
    band_softmax_kernel<<<SEQ, blk, 0, stream>>>(Qa, Ka, att, ldatt, SEQ, DAUD, SEQ);

    // 4) y = att^T @ V (banded)
    att_t_v_kernel<<<SEQ, blk, 0, stream>>>(att, ldatt, 0,   Vv, Yv, DVID, SEQ);
    att_t_v_kernel<<<SEQ, blk, 0, stream>>>(att, ldatt, SEQ, Va, Ya, DAUD, SEQ);

    // 5) output projections (Qv/Qa regions reused for results)
    f32_to_bf16_kernel<<<1024, blk, 0, stream>>>(Yv, yv_bf, (size_t)SEQ * DVID);
    f32_to_bf16_kernel<<<256,  blk, 0, stream>>>(Ya, ya_bf, (size_t)SEQ * DAUD);
    gemm_bf16_wmma_kernel<<<gV, blk, 0, stream>>>(yv_bf, wov_bf, Yv2, SEQ, DVID, DVID, 1.0f, nullptr, 0);
    gemm_bf16_wmma_kernel<<<gA, blk, 0, stream>>>(ya_bf, woa_bf, Ya2, SEQ, DAUD, DAUD, 1.0f, nullptr, 0);

    // 6) residual + LN, writing bf16 directly into concatenated [SEQ,1152] MLP input
    ln_kernel<<<SEQ, blk, 0, stream>>>(Yv2, x,  ln_y_g,  ln_y_b,  nullptr, 0, ycomb_bf, DCOMB, 0,    DVID);
    ln_kernel<<<SEQ, blk, 0, stream>>>(Ya2, xa, ln_ya_g, ln_ya_b, nullptr, 0, ycomb_bf, DCOMB, DVID, DAUD);

    // 7) MLP: h = relu(y_comb @ ka_w^T + ka_b);  h = LN(h);  score = sigmoid(h @ kd_w + kd_b)
    gemm_bf16_wmma_kernel<<<gV, blk, 0, stream>>>(ycomb_bf, kaw_bf, H, SEQ, DHID, DCOMB, 1.0f, ka_b, 1);
    ln_kernel<<<SEQ, blk, 0, stream>>>(H, nullptr, ln_ka_g, ln_ka_b, HLn, DHID, nullptr, 0, 0, DHID);
    score_kernel<<<SEQ, blk, 0, stream>>>(HLn, kd_w, kd_b, out, DHID);
}
